// BeamSearcher_12395275616619
// MI455X (gfx1250) — matrix operand order repaired and checked by reading, not compiled, not run on hardware
//
#include <hip/hip_runtime.h>
#include <hip/hip_bf16.h>
#include <math.h>

typedef __bf16 bf16;
typedef bf16  v16bf __attribute__((ext_vector_type(16)));
typedef float v8f   __attribute__((ext_vector_type(8)));

static constexpr int kV     = 5000;
static constexpr int kD     = 512;
static constexpr int kBeam  = 8;
static constexpr int kB     = 32;
static constexpr int kTenc  = 200;
static constexpr int kBB    = 256;      // kB * kBeam
static constexpr int kSteps = 100;      // T_enc * MAX_RATIO
static constexpr int kBOS   = 4998;
static constexpr int kEOS   = 4999;
static constexpr float kLenPen   = 0.5f;
static constexpr float kEosPen   = 1.5f;
static constexpr float kMinusInf = -1e20f;

static constexpr int KT1  = 32;   // 1024/32 K-tiles for [Wx;Wh]
static constexpr int NT1  = 32;   // 512/16
static constexpr int KT2  = 16;   // 512/32 K-tiles for Wo
static constexpr int NT2P = 316;  // ceil(5008/16)=313, padded to mult of 4
static constexpr int LDS2 = 5008; // scores leading dim

#define NEGINF (__int_as_float(0xff800000))

// ---------- helpers ----------
__device__ __forceinline__ bf16 f2bf(float f) {
  unsigned u = __float_as_uint(f);
  unsigned r = u + 0x7FFFu + ((u >> 16) & 1u);     // RNE
  union { unsigned short s; bf16 b; } cv;
  cv.s = (unsigned short)(r >> 16);
  return cv.b;
}

__device__ __forceinline__ v8f wmma_bf16(v16bf a, v16bf b, v8f c) {
  return __builtin_amdgcn_wmma_f32_16x16x32_bf16(false, a, false, b, (short)0, c,
                                                 false, false);
}

// A fragment (16x32 bf16, row-major src): lanes 0-15 row m0+lane K={k0..k0+7, k0+16..k0+23},
// lanes 16-31 row m0+lane-16, K shifted by +8.  Two aligned b128 loads per lane.
__device__ __forceinline__ v16bf load_a(const bf16* __restrict__ base, int ld,
                                        int m0, int k0, int lane) {
  int m = m0 + (lane & 15);
  const bf16* p = base + (size_t)m * ld + k0 + ((lane >> 4) << 3);
  union { v16bf v; uint4 q[2]; } u;
  u.q[0] = *(const uint4*)(p);       // halves 0..7
  u.q[1] = *(const uint4*)(p + 16);  // halves 16..23
  return u.v;
}

// ---------- prep kernels ----------
// swizzle [Wx;Wh] (1024x512) into B-fragment layout, bf16
__global__ void k_swz_wcat(const float* __restrict__ Wx, const float* __restrict__ Wh,
                           bf16* __restrict__ out) {
  const int total = NT1 * KT1 * 512;
  for (int i = blockIdx.x * blockDim.x + threadIdx.x; i < total;
       i += gridDim.x * blockDim.x) {
    int tile = i >> 9, w = i & 511;
    int lane = w >> 4, j = w & 15;
    int kt = tile % KT1, nt = tile / KT1;
    int k = kt * 32 + ((lane >= 16) ? 8 : 0) + j + ((j >= 8) ? 8 : 0);
    int n = nt * 16 + (lane & 15);
    float v = (k < 512) ? Wx[(size_t)k * 512 + n] : Wh[(size_t)(k - 512) * 512 + n];
    out[i] = f2bf(v);
  }
}

// swizzle Wo (512x5000, padded N to 5056) into B-fragment layout, bf16
__global__ void k_swz_wo(const float* __restrict__ Wo, bf16* __restrict__ out) {
  const size_t total = (size_t)NT2P * KT2 * 512;
  for (size_t i = blockIdx.x * (size_t)blockDim.x + threadIdx.x; i < total;
       i += (size_t)gridDim.x * blockDim.x) {
    int tile = (int)(i >> 9), w = (int)(i & 511);
    int lane = w >> 4, j = w & 15;
    int kt = tile % KT2, nt = tile / KT2;
    int k = kt * 32 + ((lane >= 16) ? 8 : 0) + j + ((j >= 8) ? 8 : 0);
    int n = nt * 16 + (lane & 15);
    float v = (n < kV) ? Wo[(size_t)k * kV + n] : 0.0f;
    out[i] = f2bf(v);
  }
}

__global__ void k_emb_bf(const float* __restrict__ Emb, bf16* __restrict__ out) {
  const size_t total = (size_t)kV * kD;
  for (size_t i = blockIdx.x * (size_t)blockDim.x + threadIdx.x; i < total;
       i += (size_t)gridDim.x * blockDim.x)
    out[i] = f2bf(Emb[i]);
}

// masked mean of encoder states -> ctx (kB x kD)
__global__ void k_ctx(const float* __restrict__ enc, const float* __restrict__ wav,
                      float* __restrict__ ctx) {
  int b = blockIdx.x, tid = threadIdx.x;
  int len = (int)rintf((float)kTenc * wav[b]);     // round-half-even like jnp.round
  if (len > kTenc) len = kTenc;
  float inv = 1.0f / (float)(len < 1 ? 1 : len);
  for (int d = tid; d < kD; d += blockDim.x) {
    float s = 0.f;
    const float* p = enc + ((size_t)b * kTenc) * kD + d;
    for (int t = 0; t < len; ++t) s += p[(size_t)t * kD];
    ctx[b * kD + d] = s * inv;
  }
}

// ctx_proj = ctx @ Wc + b   (kB x kD)
__global__ void k_ctxproj(const float* __restrict__ ctx, const float* __restrict__ Wc,
                          const float* __restrict__ bvec, float* __restrict__ ctxp) {
  int b = blockIdx.x;
  int n = blockIdx.y * blockDim.x + threadIdx.x;
  float acc = bvec[n];
  const float* c = ctx + b * kD;
  for (int k = 0; k < kD; ++k) acc += c[k] * Wc[(size_t)k * kD + n];
  ctxp[b * kD + n] = acc;
}

__global__ void k_init(float* seqsc, int* inp, float* h0, int* alivedA,
                       float* f_sc, int* f_sq, int* f_cnt) {
  int i0 = blockIdx.x * blockDim.x + threadIdx.x;
  int st = gridDim.x * blockDim.x;
  for (int i = i0; i < kBB * kD; i += st) h0[i] = 0.f;
  for (int i = i0; i < kBB; i += st) { seqsc[i] = ((i & 7) == 0) ? 0.f : NEGINF; inp[i] = kBOS; }
  for (int i = i0; i < kBB * kSteps; i += st) alivedA[i] = kEOS;
  for (int i = i0; i < kB * kBeam; i += st) f_sc[i] = NEGINF;
  for (int i = i0; i < kB * kBeam * kSteps; i += st) f_sq[i] = kEOS;
  for (int i = i0; i < kB; i += st) f_cnt[i] = 0;
}

// ---------- per-step kernels ----------
// xcat = [ bf16(Emb[inp]) | bf16(h) ]  (kBB x 1024)
__global__ void k_xcat(const int* __restrict__ inp, const bf16* __restrict__ embb,
                       const float* __restrict__ h0, bf16* __restrict__ xcat) {
  int row = blockIdx.x, tid = threadIdx.x;
  int tok = inp[row];
  const bf16* er = embb + (size_t)tok * kD;
  const float* hr = h0 + (size_t)row * kD;
  bf16* xr = xcat + (size_t)row * 1024;
  for (int c = tid; c < kD; c += blockDim.x) xr[c] = er[c];
  for (int c = tid; c < kD; c += blockDim.x) xr[kD + c] = f2bf(hr[c]);
}

// h = tanh(xcat @ [Wx;Wh] + ctx_proj)   -> hnew (f32), hbf (bf16)
__global__ void __launch_bounds__(256)
k_gemm1(const bf16* __restrict__ xcat, const bf16* __restrict__ wcat,
        const float* __restrict__ ctxp, float* __restrict__ hnew,
        bf16* __restrict__ hbf) {
  int wid = (blockIdx.x << 3) + (threadIdx.x >> 5);
  int lane = threadIdx.x & 31;
  int mt = wid >> 5, nt = wid & 31;
  v8f acc = {0.f, 0.f, 0.f, 0.f, 0.f, 0.f, 0.f, 0.f};
  int m = mt * 16 + (lane & 15);
  const bf16* arow = xcat + (size_t)m * 1024 + ((lane >> 4) << 3);
  const bf16* bbase = wcat + ((size_t)nt * KT1 << 9) + (lane << 4);
#pragma unroll 4
  for (int kt = 0; kt < KT1; ++kt) {
    union { v16bf v; uint4 q[2]; } au, bu;
    const uint4* ap = (const uint4*)(arow + kt * 32);
    au.q[0] = ap[0]; au.q[1] = ap[2];
    const uint4* bp = (const uint4*)(bbase + ((size_t)kt << 9));
    bu.q[0] = bp[0]; bu.q[1] = bp[1];
    if (kt + 1 < KT1) __builtin_prefetch((const void*)(bbase + ((size_t)(kt + 1) << 9)), 0, 1);
    acc = wmma_bf16(au.v, bu.v, acc);
  }
  int n = nt * 16 + (lane & 15);
  int mbase = mt * 16 + ((lane >> 4) << 3);
#pragma unroll
  for (int r = 0; r < 8; ++r) {
    int row = mbase + r;
    float hv = tanhf(acc[r] + ctxp[(row >> 3) * kD + n]);
    hnew[(size_t)row * kD + n] = hv;
    hbf[(size_t)row * kD + n] = f2bf(hv);
  }
}

// logits = hbf @ Wo  (kBB x 5008).
// One block per 4-N-tile group (79 blocks, 8 waves). Per K-step the 8 waves
// cooperatively stage the 4KB of B fragments into LDS with async global->LDS
// copies (ASYNCcnt), then every wave consumes them via ds_load for its two
// M-tiles: B is fetched from L2 exactly once per step instead of 16x.
__global__ void __launch_bounds__(256)
k_gemm2(const bf16* __restrict__ hbf, const bf16* __restrict__ wo,
        float* __restrict__ scores) {
  __shared__ __align__(16) uint4 sB4[256];   // 4 tiles * 1KB
  int ng   = blockIdx.x;                     // 0..78
  int w    = threadIdx.x >> 5;               // wave 0..7
  int lane = threadIdx.x & 31;

  v8f zero = {0.f, 0.f, 0.f, 0.f, 0.f, 0.f, 0.f, 0.f};
  v8f acc[2][4] = {{zero, zero, zero, zero}, {zero, zero, zero, zero}};

  // A row pointers for this wave's two M-tiles (mt = w and w+8)
  int mlo = (lane & 15);
  const bf16* arow0 = hbf + (size_t)(w * 16 + mlo) * kD + ((lane >> 4) << 3);
  const bf16* arow1 = arow0 + (size_t)128 * kD;

  // staging assignment: wave w copies 512B chunk w of the 4KB B block
  int jst  = w >> 1;                         // tile 0..3
  int half = w & 1;
  const bf16* gsrc0 = wo + ((size_t)((ng * 4 + jst) * KT2) << 9) +
                      (half << 8) + (lane << 3);
  unsigned ldsbase = (unsigned)(size_t)(&sB4[0]);
  unsigned ldsdst  = ldsbase + (unsigned)(w << 9) + (unsigned)(lane << 4);

  for (int kt = 0; kt < KT2; ++kt) {
    __syncthreads();                         // LDS free from previous iteration
    const bf16* gsrc = gsrc0 + ((size_t)kt << 9);
    asm volatile("global_load_async_to_lds_b128 %0, %1, off"
                 :: "v"(ldsdst), "v"(gsrc) : "memory");
    asm volatile("s_wait_asynccnt 0x0" ::: "memory");
    __syncthreads();                         // all 4KB staged

    union { v16bf v; uint4 q[2]; } au0, au1;
    const uint4* ap0 = (const uint4*)(arow0 + kt * 32);
    const uint4* ap1 = (const uint4*)(arow1 + kt * 32);
    au0.q[0] = ap0[0]; au0.q[1] = ap0[2];
    au1.q[0] = ap1[0]; au1.q[1] = ap1[2];
#pragma unroll
    for (int j = 0; j < 4; ++j) {
      union { v16bf v; uint4 q[2]; } bu;
      bu.q[0] = sB4[j * 64 + lane * 2];
      bu.q[1] = sB4[j * 64 + lane * 2 + 1];
      acc[0][j] = wmma_bf16(au0.v, bu.v, acc[0][j]);
      acc[1][j] = wmma_bf16(au1.v, bu.v, acc[1][j]);
    }
  }

  int ncol = lane & 15;
#pragma unroll
  for (int i = 0; i < 2; ++i) {
    int mbase = (w + i * 8) * 16 + ((lane >> 4) << 3);
#pragma unroll
    for (int j = 0; j < 4; ++j) {
      int n0 = (ng * 4 + j) * 16;
      if (n0 >= LDS2) continue;
      int n = n0 + ncol;
#pragma unroll
      for (int r = 0; r < 8; ++r)
        scores[(size_t)(mbase + r) * LDS2 + n] = acc[i][j][r];
    }
  }
}

// in-place: logits -> seq_scores + log_softmax with EOS penalty logic
__global__ void k_softmax(float* __restrict__ sc, const float* __restrict__ seqsc) {
  __shared__ float r1[256], r2[256];
  int row = blockIdx.x, tid = threadIdx.x;
  float* lg = sc + (size_t)row * LDS2;
  float m = NEGINF, mne = NEGINF;
  for (int v = tid; v < kV; v += 256) {
    float x = lg[v];
    m = fmaxf(m, x);
    if (v != kEOS) mne = fmaxf(mne, x);
  }
  r1[tid] = m; r2[tid] = mne; __syncthreads();
  for (int off = 128; off > 0; off >>= 1) {
    if (tid < off) { r1[tid] = fmaxf(r1[tid], r1[tid + off]); r2[tid] = fmaxf(r2[tid], r2[tid + off]); }
    __syncthreads();
  }
  float rowmax = r1[0], rowmaxne = r2[0];
  __syncthreads();
  float s = 0.f;
  for (int v = tid; v < kV; v += 256) s += expf(lg[v] - rowmax);
  r1[tid] = s; __syncthreads();
  for (int off = 128; off > 0; off >>= 1) {
    if (tid < off) r1[tid] += r1[tid + off];
    __syncthreads();
  }
  float lse = logf(r1[0]) + rowmax;
  float ss = seqsc[row];
  float eosp = lg[kEOS] - lse;
  float maxprobs = rowmaxne - lse;
  float eos_sc = (eosp > kEosPen * maxprobs) ? eosp : kMinusInf;
  __syncthreads();
  for (int v = tid; v < LDS2; v += 256) {
    float o;
    if (v >= kV)        o = NEGINF;
    else if (v == kEOS) o = ss + eos_sc;
    else                o = ss + (lg[v] - lse);
    lg[v] = o;
  }
}

// top-8 per batch over beam*V candidates; ties resolved to lowest index (jax order)
__global__ void k_topk(const float* __restrict__ scores, float* __restrict__ topsc,
                       int* __restrict__ cand) {
  __shared__ float sv[256];
  __shared__ int si[256];
  __shared__ int chosen[8];
  int b = blockIdx.x, tid = threadIdx.x;
  for (int pass = 0; pass < 8; ++pass) {
    float best = NEGINF; int bgi = 0x7FFFFFFF;
    for (int i = tid; i < kBeam * kV; i += 256) {
      int beam = i / kV, v = i - beam * kV;
      bool skip = false;
      for (int c = 0; c < pass; ++c) if (chosen[c] == i) skip = true;
      if (skip) continue;
      float s = scores[(size_t)(b * 8 + beam) * LDS2 + v];
      if (s > best || (s == best && i < bgi)) { best = s; bgi = i; }
    }
    sv[tid] = best; si[tid] = bgi; __syncthreads();
    for (int off = 128; off > 0; off >>= 1) {
      if (tid < off) {
        float ov = sv[tid + off]; int oi = si[tid + off];
        if (ov > sv[tid] || (ov == sv[tid] && oi < si[tid])) { sv[tid] = ov; si[tid] = oi; }
      }
      __syncthreads();
    }
    if (tid == 0) { topsc[b * 8 + pass] = sv[0]; cand[b * 8 + pass] = si[0]; chosen[pass] = si[0]; }
    __syncthreads();
  }
}

// beam permutation + EOS bookkeeping
__global__ void k_update(const float* __restrict__ topsc, const int* __restrict__ cand,
                         const float* __restrict__ hnew, float* __restrict__ h0,
                         const int* __restrict__ aA, int* __restrict__ aB,
                         float* __restrict__ f_sc, int* __restrict__ f_sq,
                         int* __restrict__ f_cnt, float* __restrict__ seqsc,
                         int* __restrict__ inp, int t) {
  int b = blockIdx.x, tid = threadIdx.x;
  __shared__ int s_tok[8], s_src[8], s_slot[8], s_valid[8];
  if (tid < 8) {
    int c = cand[b * 8 + tid];
    s_tok[tid] = c % kV;
    s_src[tid] = b * 8 + (c / kV);
  }
  __syncthreads();
  if (tid == 0) {
    int cnt = f_cnt[b], r = -1, add = 0;
    for (int j = 0; j < 8; ++j) {
      s_valid[j] = 0; s_slot[j] = 0;
      if (s_tok[j] == kEOS) {
        ++r;
        int slot = cnt + r;
        if (slot < kBeam) { s_valid[j] = 1; s_slot[j] = slot; ++add; }
      }
    }
    f_cnt[b] = cnt + add;
  }
  __syncthreads();
  if (tid < 8) {
    int row = b * 8 + tid;
    int tok = s_tok[tid];
    float sc = topsc[b * 8 + tid];
    inp[row] = tok;
    seqsc[row] = (tok == kEOS) ? NEGINF : sc;
    if (s_valid[tid]) f_sc[b * 8 + s_slot[tid]] = sc + kLenPen * (float)(t + 1);
  }
  __syncthreads();
  for (int idx = tid; idx < 8 * kD; idx += blockDim.x) {
    int j = idx >> 9, d = idx & 511;
    h0[(size_t)(b * 8 + j) * kD + d] = hnew[(size_t)s_src[j] * kD + d];
  }
  for (int idx = tid; idx < 8 * kSteps; idx += blockDim.x) {
    int j = idx / kSteps, s = idx % kSteps;
    int val = (s == t) ? s_tok[j] : aA[s_src[j] * kSteps + s];
    aB[(b * 8 + j) * kSteps + s] = val;
    if (s_valid[j]) f_sq[(b * 8 + s_slot[j]) * kSteps + s] = val;
  }
}

__global__ void k_finalize(float* __restrict__ f_sc, int* __restrict__ f_sq,
                           const int* __restrict__ f_cnt, const float* __restrict__ seqsc,
                           const int* __restrict__ alived, float* __restrict__ out) {
  int b = blockIdx.x, tid = threadIdx.x;
  __shared__ float fs[8];
  __shared__ int s_best, s_len, s_cnt;
  if (tid == 0) s_cnt = f_cnt[b];
  __syncthreads();
  int cnt = s_cnt;
  if (tid < 8) {
    int j = tid, src = j - cnt;
    src = src < 0 ? 0 : (src > 7 ? 7 : src);
    fs[j] = (j >= cnt) ? (seqsc[b * 8 + src] + kLenPen * (float)kSteps) : f_sc[b * 8 + j];
  }
  __syncthreads();
  for (int idx = tid; idx < 8 * kSteps; idx += blockDim.x) {
    int j = idx / kSteps, s = idx % kSteps;
    if (j >= cnt) {
      int src = j - cnt; src = src < 0 ? 0 : (src > 7 ? 7 : src);
      f_sq[(b * 8 + j) * kSteps + s] = alived[(b * 8 + src) * kSteps + s];
    }
  }
  __syncthreads();
  if (tid == 0) {
    int best = 0; float bv = fs[0];
    for (int j = 1; j < 8; ++j) if (fs[j] > bv) { bv = fs[j]; best = j; }
    s_best = best;
    out[kB * kSteps + kB + b] = bv;                 // top_scores
  }
  __syncthreads();
  int best = s_best;
  if (tid == 0) {
    const int* pr = f_sq + (b * 8 + best) * kSteps;
    int len = 0;
    while (len < kSteps && pr[len] != kEOS) ++len;
    s_len = len;
    out[kB * kSteps + b] = (float)len;              // lengths
  }
  __syncthreads();
  int len = s_len;
  const int* pr = f_sq + (b * 8 + best) * kSteps;
  for (int s = tid; s < kSteps; s += blockDim.x)
    out[b * kSteps + s] = (float)((s < len) ? pr[s] : kEOS);   // preds
}

// ---------- host ----------
extern "C" void kernel_launch(void* const* d_in, const int* in_sizes, int n_in,
                              void* d_out, int out_size, void* d_ws, size_t ws_size,
                              hipStream_t stream) {
  (void)in_sizes; (void)n_in; (void)out_size; (void)ws_size;
  const float* enc = (const float*)d_in[0];
  const float* wav = (const float*)d_in[1];
  const float* Emb = (const float*)d_in[2];
  const float* Wx  = (const float*)d_in[3];
  const float* Wh  = (const float*)d_in[4];
  const float* Wc  = (const float*)d_in[5];
  const float* bv  = (const float*)d_in[6];
  const float* Wo  = (const float*)d_in[7];
  float* out = (float*)d_out;

  char* ws = (char*)d_ws;
  size_t off = 0;
  auto carve = [&](size_t bytes) -> char* {
    char* p = ws + off;
    off = (off + bytes + 255) & ~(size_t)255;
    return p;
  };
  bf16*  wcat  = (bf16*) carve((size_t)NT1 * KT1 * 512 * 2);
  bf16*  wo    = (bf16*) carve((size_t)NT2P * KT2 * 512 * 2);
  bf16*  embb  = (bf16*) carve((size_t)kV * kD * 2);
  bf16*  xcat  = (bf16*) carve((size_t)kBB * 1024 * 2);
  float* h0    = (float*)carve((size_t)kBB * kD * 4);
  float* hnew  = (float*)carve((size_t)kBB * kD * 4);
  bf16*  hbf   = (bf16*) carve((size_t)kBB * kD * 2);
  float* ctx   = (float*)carve((size_t)kB * kD * 4);
  float* ctxp  = (float*)carve((size_t)kB * kD * 4);
  float* score = (float*)carve((size_t)kBB * LDS2 * 4);
  float* seqsc = (float*)carve((size_t)kBB * 4);
  int*   inp   = (int*)  carve((size_t)kBB * 4);
  float* topsc = (float*)carve((size_t)kB * kBeam * 4);
  int*   cand  = (int*)  carve((size_t)kB * kBeam * 4);
  int*   aA    = (int*)  carve((size_t)kBB * kSteps * 4);
  int*   aB    = (int*)  carve((size_t)kBB * kSteps * 4);
  float* fsc   = (float*)carve((size_t)kB * kBeam * 4);
  int*   fsq   = (int*)  carve((size_t)kB * kBeam * kSteps * 4);
  int*   fcnt  = (int*)  carve((size_t)kB * 4);

  // one-time prep (re-run every call: deterministic)
  k_swz_wcat<<<512, 256, 0, stream>>>(Wx, Wh, wcat);
  k_swz_wo  <<<1024, 256, 0, stream>>>(Wo, wo);
  k_emb_bf  <<<1024, 256, 0, stream>>>(Emb, embb);
  k_ctx     <<<kB, 256, 0, stream>>>(enc, wav, ctx);
  k_ctxproj <<<dim3(kB, 4), 128, 0, stream>>>(ctx, Wc, bv, ctxp);
  k_init    <<<256, 256, 0, stream>>>(seqsc, inp, h0, aA, fsc, fsq, fcnt);

  int* pa = aA;
  int* pb = aB;
  for (int t = 0; t < kSteps; ++t) {
    k_xcat   <<<kBB, 256, 0, stream>>>(inp, embb, h0, xcat);
    k_gemm1  <<<64, 256, 0, stream>>>(xcat, wcat, ctxp, hnew, hbf);
    k_gemm2  <<<79, 256, 0, stream>>>(hbf, wo, score);   // 8 waves/block, LDS-staged B
    k_softmax<<<kBB, 256, 0, stream>>>(score, seqsc);
    k_topk   <<<kB, 256, 0, stream>>>(score, topsc, cand);
    k_update <<<kB, 128, 0, stream>>>(topsc, cand, hnew, h0, pa, pb,
                                      fsc, fsq, fcnt, seqsc, inp, t);
    int* tmp = pa; pa = pb; pb = tmp;
  }
  k_finalize<<<kB, 128, 0, stream>>>(fsc, fsq, fcnt, seqsc, pa, out);
}
